// HyperLSTMCell_81552839017160
// MI455X (gfx1250) — matrix operand order repaired and checked by reading, not compiled, not run on hardware
//
#include <hip/hip_runtime.h>

// ---------------------------------------------------------------------------
// Types for CDNA5 WMMA
// ---------------------------------------------------------------------------
typedef __bf16 v16bf __attribute__((ext_vector_type(16)));
typedef float  v8f   __attribute__((ext_vector_type(8)));
typedef unsigned int u32x4 __attribute__((ext_vector_type(4)));
typedef int v4i __attribute__((ext_vector_type(4)));

union Frag { u32x4 q[2]; v16bf v; };

#define B_   4096
#define IN_  1024
#define H_   2048
#define HU_  128
#define E_   16
#define EPS_ 0.001f
#define FB_  1.0f

#define GLOBAL_AS __attribute__((address_space(1)))
#define SHARED_AS __attribute__((address_space(3)))

#if defined(__gfx1250__) && __has_builtin(__builtin_amdgcn_global_load_async_to_lds_b128) && __has_builtin(__builtin_amdgcn_s_wait_asynccnt)
#define USE_ASYNC 1
#else
#define USE_ASYNC 0
#endif

static constexpr size_t O1 = (size_t)B_ * H_;                 // new_total_c offset
static constexpr size_t O2 = O1 + (size_t)B_ * (H_ + HU_);    // new_total_h offset

__device__ __forceinline__ unsigned short f2bf(float f) {
    unsigned int u = __float_as_uint(f);
    u += 0x7FFFu + ((u >> 16) & 1u);          // round-to-nearest-even
    return (unsigned short)(u >> 16);
}
__device__ __forceinline__ float sigm(float x) { return 1.0f / (1.0f + expf(-x)); }

template<int NT>
__device__ __forceinline__ float bred(float x, float* red) {
    int t = threadIdx.x;
    red[t] = x; __syncthreads();
    #pragma unroll
    for (int s = NT / 2; s > 0; s >>= 1) {
        if (t < s) red[t] += red[t + s];
        __syncthreads();
    }
    float r = red[0]; __syncthreads();
    return r;
}

// ---------------------------------------------------------------------------
// 1) Pack activations: A = bf16([x | h | hyper_h])   (4096 x 3200)
// ---------------------------------------------------------------------------
__global__ __launch_bounds__(256) void k_pack_A(const float* __restrict__ x,
                                                const float* __restrict__ th,
                                                unsigned short* __restrict__ A) {
    int i = blockIdx.x * 256 + threadIdx.x;
    if (i >= B_ * 3200) return;
    int b = i / 3200, k = i % 3200;
    float v;
    if (k < IN_)            v = x[b * IN_ + k];
    else if (k < IN_ + H_)  v = th[b * (H_ + HU_) + (k - IN_)];
    else                    v = th[b * (H_ + HU_) + H_ + (k - IN_ - H_)];
    A[i] = f2bf(v);
}

// ---------------------------------------------------------------------------
// 2) Transpose-convert weights: out[n][kofs+k] = bf16(in[k][n]); in is K x N
// ---------------------------------------------------------------------------
__global__ __launch_bounds__(256) void k_pack_Wt(const float* __restrict__ in, int K, int N,
                                                 unsigned short* __restrict__ out, int ldo, int kofs) {
    __shared__ float tile[32][33];
    int bk = blockIdx.y * 32, bn = blockIdx.x * 32;
    for (int r = threadIdx.y; r < 32; r += 8)
        tile[r][threadIdx.x] = in[(size_t)(bk + r) * N + bn + threadIdx.x];
    __syncthreads();
    for (int r = threadIdx.y; r < 32; r += 8)
        out[(size_t)(bn + r) * ldo + kofs + bk + threadIdx.x] = f2bf(tile[threadIdx.x][r]);
}

// ---------------------------------------------------------------------------
// 3) bf16 WMMA GEMM: C[M][N] = A[:, aofs:aofs+K] * Bt[N][K]^T
//    128x128 tile / block, 8 waves, each wave 32x64 (2x4 wmma accumulators).
//    Data path: GLOBAL_LOAD_ASYNC_TO_LDS_B128 double-buffered (ASYNCcnt),
//    falling back to register-staged global->LDS if the builtin is absent.
// ---------------------------------------------------------------------------
#define LDK 40   // LDS row stride in bf16 (80 bytes = 5 * 16B: keeps b128 alignment)

__global__ __launch_bounds__(256) void k_gemm_bf16(
        const unsigned short* __restrict__ A, int lda, int aofs,
        const unsigned short* __restrict__ Bt, int ldb,
        float* __restrict__ C, int ldc, int K) {
#if USE_ASYNC
    __shared__ __align__(16) unsigned short sA[2][128 * LDK];
    __shared__ __align__(16) unsigned short sB[2][128 * LDK];
#else
    __shared__ __align__(16) unsigned short sA[1][128 * LDK];
    __shared__ __align__(16) unsigned short sB[1][128 * LDK];
#endif

    const int t    = threadIdx.x;
    const int lane = t & 31, wid = t >> 5;
    const int wM = (wid >> 1) * 32;     // wave M origin within tile (0,32,64,96)
    const int wN = (wid & 1) * 64;      // wave N origin within tile (0,64)
    const int half = lane >> 4, mrow = lane & 15;
    const int rowA = t >> 1, hlf = t & 1;    // 256 thr -> 128 rows x 2 halves (16 bf16 = 32B)
    const int m0 = blockIdx.y * 128, n0 = blockIdx.x * 128;

    const unsigned short* gA = A  + (size_t)(m0 + rowA) * lda + aofs + hlf * 16;
    const unsigned short* gB = Bt + (size_t)(n0 + rowA) * ldb + hlf * 16;
    const int stOff = rowA * LDK + hlf * 16;

    v8f acc[2][4];
    #pragma unroll
    for (int i = 0; i < 2; ++i)
        #pragma unroll
        for (int j = 0; j < 4; ++j)
            acc[i][j] = (v8f){0.f,0.f,0.f,0.f,0.f,0.f,0.f,0.f};

    const int steps = K >> 5;

#if USE_ASYNC
    // Direct-to-LDS async copy (no VGPR staging); 4 b128 issues per step.
    auto issue = [&](int buf, int s) {
        GLOBAL_AS v4i* ga = (GLOBAL_AS v4i*)(gA + (size_t)s * 32);
        GLOBAL_AS v4i* gb = (GLOBAL_AS v4i*)(gB + (size_t)s * 32);
        SHARED_AS v4i* la = (SHARED_AS v4i*)&sA[buf][stOff];
        SHARED_AS v4i* lb = (SHARED_AS v4i*)&sB[buf][stOff];
        __builtin_amdgcn_global_load_async_to_lds_b128(ga,     la,     0, 0);
        __builtin_amdgcn_global_load_async_to_lds_b128(ga + 1, la + 1, 0, 0);
        __builtin_amdgcn_global_load_async_to_lds_b128(gb,     lb,     0, 0);
        __builtin_amdgcn_global_load_async_to_lds_b128(gb + 1, lb + 1, 0, 0);
    };
    issue(0, 0);
    for (int s = 0; s < steps; ++s) {
        const int buf = s & 1;
        if (s + 1 < steps) {
            issue(buf ^ 1, s + 1);
            // async loads complete in order: allow next tile's 4 to stay in flight
            __builtin_amdgcn_s_wait_asynccnt(4);
        } else {
            __builtin_amdgcn_s_wait_asynccnt(0);
        }
        __syncthreads();
        const unsigned short* bA = sA[buf];
        const unsigned short* bB = sB[buf];
#else
    u32x4 ra0 = *(const u32x4*)gA,       ra1 = *(const u32x4*)(gA + 8);
    u32x4 rb0 = *(const u32x4*)gB,       rb1 = *(const u32x4*)(gB + 8);
    for (int s = 0; s < steps; ++s) {
        *(u32x4*)&sA[0][stOff]     = ra0;
        *(u32x4*)&sA[0][stOff + 8] = ra1;
        *(u32x4*)&sB[0][stOff]     = rb0;
        *(u32x4*)&sB[0][stOff + 8] = rb1;
        __syncthreads();
        if (s + 1 < steps) {
            const unsigned short* na = gA + (size_t)(s + 1) * 32;
            const unsigned short* nb = gB + (size_t)(s + 1) * 32;
            ra0 = *(const u32x4*)na;  ra1 = *(const u32x4*)(na + 8);
            rb0 = *(const u32x4*)nb;  rb1 = *(const u32x4*)(nb + 8);
            if (s + 2 < steps) {
                __builtin_prefetch(gA + (size_t)(s + 2) * 32, 0, 1);
                __builtin_prefetch(gB + (size_t)(s + 2) * 32, 0, 1);
            }
        }
        const unsigned short* bA = sA[0];
        const unsigned short* bB = sB[0];
#endif
        Frag a[2], b[4];
        // A frag (ISA 16-bit A 16x32): lanes<16: K 0..7 & 16..23; lanes>=16: K 8..15 & 24..31
        #pragma unroll
        for (int fm = 0; fm < 2; ++fm) {
            const unsigned short* p = &bA[(wM + fm * 16 + mrow) * LDK];
            a[fm].q[0] = *(const u32x4*)(p + half * 8);
            a[fm].q[1] = *(const u32x4*)(p + 16 + half * 8);
        }
        // B frag (same K order as A): lanes<16: K 0..7 & 16..23; lanes>=16: K 8..15 & 24..31, n = lane%16
        #pragma unroll
        for (int fn = 0; fn < 4; ++fn) {
            const unsigned short* p = &bB[(wN + fn * 16 + mrow) * LDK];
            b[fn].q[0] = *(const u32x4*)(p + half * 8);
            b[fn].q[1] = *(const u32x4*)(p + 16 + half * 8);
        }
        #pragma unroll
        for (int fm = 0; fm < 2; ++fm)
            #pragma unroll
            for (int fn = 0; fn < 4; ++fn)
                acc[fm][fn] = __builtin_amdgcn_wmma_f32_16x16x32_bf16(
                    false, a[fm].v, false, b[fn].v, (short)0, acc[fm][fn], false, false);
        __syncthreads();
    }

    // C layout: VGPR r -> M = half*8 + r, N = lane%16
    #pragma unroll
    for (int fm = 0; fm < 2; ++fm) {
        int m = m0 + wM + fm * 16 + half * 8;
        #pragma unroll
        for (int fn = 0; fn < 4; ++fn) {
            int n = n0 + wN + fn * 16 + mrow;
            float* cp = C + (size_t)m * ldc + n;
            #pragma unroll
            for (int r = 0; r < 8; ++r) cp[(size_t)r * ldc] = acc[fm][fn][r];
        }
    }
}

// ---------------------------------------------------------------------------
// 4) Hyper LSTM activations: LN(4x128) -> gates -> new_hyper_c/h
// ---------------------------------------------------------------------------
__global__ __launch_bounds__(128) void k_hyper_act(
        const float* __restrict__ HC, const float* __restrict__ h_bias,
        const float* __restrict__ gma, const float* __restrict__ bta,
        const float* __restrict__ cg,  const float* __restrict__ cb,
        const float* __restrict__ total_c,
        float* __restrict__ out_c, float* __restrict__ out_h,
        float* __restrict__ hyper_out) {
    __shared__ float red[128];
    int b = blockIdx.x, t = threadIdx.x;
    float v[4], nrm[4];
    #pragma unroll
    for (int g = 0; g < 4; ++g) v[g] = HC[b * 512 + g * 128 + t] + h_bias[g * 128 + t];
    for (int g = 0; g < 4; ++g) {
        float m  = bred<128>(v[g], red) * (1.0f / 128.0f);
        float d  = v[g] - m;
        float vr = bred<128>(d * d, red) * (1.0f / 128.0f);
        nrm[g] = gma[g * 128 + t] * (d * rsqrtf(vr + EPS_)) + bta[g * 128 + t];
    }
    float hi = nrm[0], hj = tanhf(nrm[1]), hf = nrm[2], ho = nrm[3];
    float c_old = total_c[(size_t)b * (H_ + HU_) + H_ + t];
    float nc = c_old * sigm(hf + FB_) + sigm(hi) * hj;
    float m  = bred<128>(nc, red) * (1.0f / 128.0f);
    float d  = nc - m;
    float vr = bred<128>(d * d, red) * (1.0f / 128.0f);
    float nh = tanhf(cg[t] * (d * rsqrtf(vr + EPS_)) + cb[t]) * sigm(ho);
    out_c[(size_t)b * (H_ + HU_) + H_ + t] = nc;
    out_h[(size_t)b * (H_ + HU_) + H_ + t] = nh;
    hyper_out[b * HU_ + t] = nh;
}

// ---------------------------------------------------------------------------
// 5) ZW[k][b][e] = hyper_out[b]·hyp_zw[k][:,e] + hyp_zb[k][e];  ZB likewise
// ---------------------------------------------------------------------------
__global__ __launch_bounds__(192) void k_zwzb(
        const float* __restrict__ hyper_out, const float* __restrict__ hyp_zw,
        const float* __restrict__ hyp_zb, const float* __restrict__ hyp_zbw,
        float* __restrict__ ZW, float* __restrict__ ZB) {
    __shared__ float ho[128];
    int b = blockIdx.x;
    if (threadIdx.x < 128) ho[threadIdx.x] = hyper_out[b * HU_ + threadIdx.x];
    __syncthreads();
    int k = threadIdx.x / 16, e = threadIdx.x % 16;
    float sw = 0.f, sb = 0.f;
    #pragma unroll 4
    for (int u = 0; u < 128; ++u) {
        float h = ho[u];
        sw += h * hyp_zw [(k * 128 + u) * 16 + e];
        sb += h * hyp_zbw[(k * 128 + u) * 16 + e];
    }
    ZW[(size_t)(k * B_ + b) * 16 + e] = sw + hyp_zb[k * 16 + e];
    ZB[(size_t)(k * B_ + b) * 16 + e] = sb;
}

// ---------------------------------------------------------------------------
// 6) Mod-sum: GP[b][g*2048+n] = sx*XH + sh*HH + sb*bias + hb (in place on XH)
// ---------------------------------------------------------------------------
__global__ __launch_bounds__(256) void k_modsum(
        const float* __restrict__ XH, const float* __restrict__ HH,
        const float* __restrict__ bias,
        const float* __restrict__ ZW, const float* __restrict__ ZB,
        const float* __restrict__ hyp_ww, const float* __restrict__ hyp_bw,
        float* __restrict__ GP) {
    __shared__ float sww[3][16][64], sbw[3][16][64];
    __shared__ float szw[3][64][16], szb[3][64][16];
    int g  = blockIdx.z;
    int b0 = blockIdx.y * 64, n0 = blockIdx.x * 64;
    int t  = threadIdx.x;
    for (int i = t; i < 3 * 16 * 64; i += 256) {
        int kk = i / 1024, e = (i % 1024) / 64, c = i % 64;
        int k = g + kk * 4;
        sww[kk][e][c] = hyp_ww[(size_t)(k * 16 + e) * H_ + n0 + c];
        sbw[kk][e][c] = hyp_bw[(size_t)(k * 16 + e) * H_ + n0 + c];
    }
    for (int i = t; i < 3 * 64 * 16; i += 256) {
        int kk = i / 1024, r = (i % 1024) / 16, e = i % 16;
        int k = g + kk * 4;
        szw[kk][r][e] = ZW[(size_t)(k * B_ + b0 + r) * 16 + e];
        szb[kk][r][e] = ZB[(size_t)(k * B_ + b0 + r) * 16 + e];
    }
    __syncthreads();
    int c = t & 63;
    #pragma unroll 4
    for (int j = 0; j < 16; ++j) {
        int r = (t >> 6) + j * 4;
        float sx = 0.f, sh = 0.f, sb2 = 0.f, hb = 0.f;
        #pragma unroll
        for (int e = 0; e < 16; ++e) {
            sx  += szw[0][r][e] * sww[0][e][c];
            sh  += szw[1][r][e] * sww[1][e][c];
            sb2 += szw[2][r][e] * sww[2][e][c];
            hb  += szb[0][r][e] * sbw[0][e][c]
                 + szb[1][r][e] * sbw[1][e][c]
                 + szb[2][r][e] * sbw[2][e][c];
        }
        int bb = b0 + r, col = g * H_ + n0 + c;
        size_t idx = (size_t)bb * (4 * H_) + col;
        float xh = XH[idx], hh = HH[idx];
        GP[idx] = sx * xh + sh * hh + sb2 * bias[col] + hb;
    }
}

// ---------------------------------------------------------------------------
// 7) Per-row: LN over 4 gate groups -> cell -> LN(new_c) -> new_h
// ---------------------------------------------------------------------------
__global__ __launch_bounds__(256) void k_ln_gates(
        const float* __restrict__ GP,
        const float* __restrict__ gma, const float* __restrict__ bta,
        const float* __restrict__ cg,  const float* __restrict__ cb,
        const float* __restrict__ total_c, float* __restrict__ out) {
    __shared__ float red[256];
    __shared__ float mu[4], rs[4];
    __shared__ float mc_s, rc_s;
    __shared__ float s_nc[H_];
    __shared__ float s_so[H_];
    int b = blockIdx.x, t = threadIdx.x;
    const float* row = GP + (size_t)b * (4 * H_);

    for (int g = 0; g < 4; ++g) {
        float s = 0.f, ss = 0.f;
        for (int i = t; i < H_; i += 256) {
            float v = row[g * H_ + i];
            s += v; ss += v * v;
        }
        s  = bred<256>(s,  red);
        ss = bred<256>(ss, red);
        if (t == 0) {
            float m = s * (1.0f / H_);
            mu[g] = m;
            rs[g] = rsqrtf(ss * (1.0f / H_) - m * m + EPS_);
        }
    }
    __syncthreads();

    float sc = 0.f, ssc = 0.f;
    for (int i = t; i < H_; i += 256) {
        float gi = gma[0 * H_ + i] * ((row[0 * H_ + i] - mu[0]) * rs[0]) + bta[0 * H_ + i];
        float gj = gma[1 * H_ + i] * ((row[1 * H_ + i] - mu[1]) * rs[1]) + bta[1 * H_ + i];
        float gf = gma[2 * H_ + i] * ((row[2 * H_ + i] - mu[2]) * rs[2]) + bta[2 * H_ + i];
        float go = gma[3 * H_ + i] * ((row[3 * H_ + i] - mu[3]) * rs[3]) + bta[3 * H_ + i];
        float c_old = total_c[(size_t)b * (H_ + HU_) + i];
        float nc = c_old * sigm(gf + FB_) + sigm(gi) * tanhf(gj);
        s_nc[i] = nc;
        s_so[i] = sigm(go);
        sc += nc; ssc += nc * nc;
    }
    sc  = bred<256>(sc,  red);
    ssc = bred<256>(ssc, red);
    if (t == 0) {
        float m = sc * (1.0f / H_);
        mc_s = m;
        rc_s = rsqrtf(ssc * (1.0f / H_) - m * m + EPS_);
    }
    __syncthreads();
    float mc = mc_s, rc = rc_s;
    for (int i = t; i < H_; i += 256) {
        float nc = s_nc[i];
        float nh = tanhf(cg[i] * ((nc - mc) * rc) + cb[i]) * s_so[i];
        out[(size_t)b * H_ + i]              = nh;   // new_h
        out[O1 + (size_t)b * (H_ + HU_) + i] = nc;   // new_total_c[:, :H]
        out[O2 + (size_t)b * (H_ + HU_) + i] = nh;   // new_total_h[:, :H]
    }
}

// ---------------------------------------------------------------------------
extern "C" void kernel_launch(void* const* d_in, const int* in_sizes, int n_in,
                              void* d_out, int out_size, void* d_ws, size_t ws_size,
                              hipStream_t stream) {
    const float* x        = (const float*)d_in[0];
    const float* total_c  = (const float*)d_in[1];
    const float* total_h  = (const float*)d_in[2];
    const float* W_xh     = (const float*)d_in[3];
    const float* W_hh     = (const float*)d_in[4];
    const float* bias     = (const float*)d_in[5];
    const float* ln_gamma = (const float*)d_in[6];
    const float* ln_beta  = (const float*)d_in[7];
    const float* ln_c_g   = (const float*)d_in[8];
    const float* ln_c_b   = (const float*)d_in[9];
    const float* hW_xh    = (const float*)d_in[10];
    const float* hW_hh    = (const float*)d_in[11];
    const float* h_bias   = (const float*)d_in[12];
    const float* h_ln_g   = (const float*)d_in[13];
    const float* h_ln_b   = (const float*)d_in[14];
    const float* h_ln_cg  = (const float*)d_in[15];
    const float* h_ln_cb  = (const float*)d_in[16];
    const float* hyp_zw   = (const float*)d_in[17];
    const float* hyp_zb   = (const float*)d_in[18];
    const float* hyp_ww   = (const float*)d_in[19];
    const float* hyp_zbw  = (const float*)d_in[20];
    const float* hyp_bw   = (const float*)d_in[21];
    float* out = (float*)d_out;

    char* ws = (char*)d_ws;
    size_t off = 0;
    auto alloc = [&](size_t bytes) -> char* {
        char* p = ws + off;
        off += (bytes + 255) & ~(size_t)255;
        return p;
    };
    unsigned short* AbfH = (unsigned short*)alloc((size_t)B_ * 3200 * 2);
    unsigned short* Wt1  = (unsigned short*)alloc((size_t)(4 * H_) * IN_ * 2);
    unsigned short* Wt2  = (unsigned short*)alloc((size_t)(4 * H_) * H_ * 2);
    unsigned short* WtH  = (unsigned short*)alloc((size_t)(4 * HU_) * 3200 * 2);
    float* HC        = (float*)alloc((size_t)B_ * 4 * HU_ * 4);
    float* hyper_out = (float*)alloc((size_t)B_ * HU_ * 4);
    float* ZW        = (float*)alloc((size_t)12 * B_ * 16 * 4);
    float* ZB        = (float*)alloc((size_t)12 * B_ * 16 * 4);
    float* XH        = (float*)alloc((size_t)B_ * 4 * H_ * 4);
    float* HH        = (float*)alloc((size_t)B_ * 4 * H_ * 4);

    dim3 b256(256);
    dim3 tb(32, 8);

    // Pack activations and weights to bf16 (K-contiguous B for WMMA frags)
    k_pack_A<<<dim3((B_ * 3200 + 255) / 256), b256, 0, stream>>>(x, total_h, AbfH);
    k_pack_Wt<<<dim3(8192 / 32, 1024 / 32), tb, 0, stream>>>(W_xh, 1024, 8192, Wt1, 1024, 0);
    k_pack_Wt<<<dim3(8192 / 32, 2048 / 32), tb, 0, stream>>>(W_hh, 2048, 8192, Wt2, 2048, 0);
    k_pack_Wt<<<dim3(512 / 32, 3072 / 32), tb, 0, stream>>>(hW_xh, 3072, 512, WtH, 3200, 0);
    k_pack_Wt<<<dim3(512 / 32, 128 / 32), tb, 0, stream>>>(hW_hh, 128, 512, WtH, 3200, 3072);

    // Hyper LSTM path
    k_gemm_bf16<<<dim3(512 / 128, B_ / 128), b256, 0, stream>>>(AbfH, 3200, 0, WtH, 3200, HC, 512, 3200);
    k_hyper_act<<<B_, 128, 0, stream>>>(HC, h_bias, h_ln_g, h_ln_b, h_ln_cg, h_ln_cb,
                                        total_c, out + O1, out + O2, hyper_out);
    k_zwzb<<<B_, 192, 0, stream>>>(hyper_out, hyp_zw, hyp_zb, hyp_zbw, ZW, ZB);

    // Main GEMMs: XH = x @ W_xh, HH = h @ W_hh (bf16 WMMA, fp32 accumulate)
    k_gemm_bf16<<<dim3(8192 / 128, B_ / 128), b256, 0, stream>>>(AbfH, 3200, 0,    Wt1, 1024, XH, 8192, 1024);
    k_gemm_bf16<<<dim3(8192 / 128, B_ / 128), b256, 0, stream>>>(AbfH, 3200, 1024, Wt2, 2048, HH, 8192, 2048);

    // Gate modulation (in place on XH) and final LN/cell
    k_modsum<<<dim3(H_ / 64, B_ / 64, 4), b256, 0, stream>>>(XH, HH, bias, ZW, ZB, hyp_ww, hyp_bw, XH);
    k_ln_gates<<<B_, 256, 0, stream>>>(XH, ln_gamma, ln_beta, ln_c_g, ln_c_b, total_c, out);
}
